// MHSA_22488448762582
// MI455X (gfx1250) — compile-verified, matrix operand order checked by reading, not run on hardware
//
#include <hip/hip_runtime.h>

// ---------------------------------------------------------------------------
// Talking-heads attention, fused for gfx1250 (MI455X, wave32, WMMA).
//   B=2, P=2048, D_MODEL=512, H=8, DH=64, SCALE=8
// Pipeline:
//   1) x -> f16; build pre-mixed Wq' (W1 & 1/SCALE folded in), transposed f16
//      weights for K/V/proj.
//   2) WMMA GEMMs: Q' = x@Wq' [4096x4096], K = x@Wk [4096x512],
//      V = x@Wv stored transposed [b][g][64][P].
//   3) Fused 2-pass attention: mixed logits via 512-deep WMMA, exact softmax,
//      cross-wave W2 mix through LDS, PV WMMA accumulation. K tiles staged
//      into LDS via GLOBAL_LOAD_ASYNC_TO_LDS_B128 (ASYNCcnt) when available.
//   4) Output projection WMMA GEMM -> f32 d_out.
// ---------------------------------------------------------------------------

typedef _Float16 f16;
typedef _Float16 v16h __attribute__((ext_vector_type(16)));
typedef _Float16 v8h  __attribute__((ext_vector_type(8)));
typedef float    v8f  __attribute__((ext_vector_type(8)));

union V16 { v16h v; v8h h2[2]; f16 e[16]; };
union F8  { v8f v; float f[8]; };

#define PSEQ 2048
#define DM   512
#define NH   8
#define DH   64

#if __has_builtin(__builtin_amdgcn_global_load_async_to_lds_b128)
#define USE_ASYNC_LDS 1
#endif

static __device__ __forceinline__ v8f wmma_f16(v16h a, v16h b, v8f c) {
  return __builtin_amdgcn_wmma_f32_16x16x32_f16(false, a, false, b, (short)0, c,
                                                false, false);
}

// Copy 16B global -> LDS. Async direct-to-LDS path on gfx1250 when available
// (no VGPR round trip; tracked by ASYNCcnt), else plain load+store.
#ifdef USE_ASYNC_LDS
typedef int v4i __attribute__((vector_size(16)));
typedef __attribute__((address_space(1))) v4i gv4i;  // global int4
typedef __attribute__((address_space(3))) v4i lv4i;  // LDS int4
#endif

static __device__ __forceinline__ void cp16_g2l(const f16* __restrict__ gsrc,
                                                f16* __restrict__ ldst) {
#ifdef USE_ASYNC_LDS
  // flat->as(1): same VA; flat->as(3): low 32 bits of a flat LDS address are
  // the LDS byte offset.
  __builtin_amdgcn_global_load_async_to_lds_b128(
      (gv4i*)(uintptr_t)gsrc, (lv4i*)(uintptr_t)ldst, 0, 0);
#else
  *(uint4*)ldst = *(const uint4*)gsrc;
#endif
}

static __device__ __forceinline__ void cp_wait() {
#ifdef USE_ASYNC_LDS
#if __has_builtin(__builtin_amdgcn_s_wait_asynccnt)
  __builtin_amdgcn_s_wait_asynccnt(0);
#else
  asm volatile("s_wait_asynccnt 0x0" ::: "memory");
#endif
#endif
}

// Load one lane's A/B operand slice for a 32-deep K chunk from a row-major row.
// ISA layout: lanes 0-15 hold K {0..7,16..23}, lanes 16-31 hold K {8..15,24..31},
// so per lane: two contiguous 16B reads at +hsel and +hsel+16 halfs.
static __device__ __forceinline__ v16h ldp(const f16* __restrict__ row, int halfoff) {
  V16 t;
  t.h2[0] = *(const v8h*)(row + halfoff);
  t.h2[1] = *(const v8h*)(row + halfoff + 16);
  return t.v;
}

// ---------------------------- prep kernels ---------------------------------

__global__ void k_cvt_f16(const float* __restrict__ s, f16* __restrict__ d, int n) {
  int i = blockIdx.x * blockDim.x + threadIdx.x;
  if (i < n) d[i] = (f16)s[i];
}

// Wq'[(g*512+h*64+d), c] = Wq[c, h*64+d] * W1[h,g] / 8   (stored [N=4096, K=512])
__global__ void k_prep_wq(const float* __restrict__ Wq, const float* __restrict__ W1,
                          f16* __restrict__ WqTb) {
  int i = blockIdx.x * blockDim.x + threadIdx.x;
  if (i >= 4096 * 512) return;
  int c = i & 511, row = i >> 9;
  int g = row >> 9, hd = row & 511, h = hd >> 6;
  WqTb[i] = (f16)(Wq[c * 512 + hd] * W1[h * 8 + g] * 0.125f);
}

// WT[n, c] = W[c, n]  (512x512)
__global__ void k_transpose512(const float* __restrict__ W, f16* __restrict__ WT) {
  int i = blockIdx.x * blockDim.x + threadIdx.x;
  if (i >= 512 * 512) return;
  int c = i & 511, n = i >> 9;
  WT[i] = (f16)W[c * 512 + n];
}

// ------------------------------ WMMA GEMM ----------------------------------
// One wave computes a 16x64 tile of C = A[M,512] @ BT[N,512]^T.
// mode 0: f16 row-major out (ldout), mode 1: f16 stored as V^T [b][g][64][P],
// mode 2: f32 row-major out (final output).
__global__ __launch_bounds__(32) void k_gemm(const f16* __restrict__ A,
                                             const f16* __restrict__ BT,
                                             void* __restrict__ out,
                                             int ldout, int mode) {
  const int lane = threadIdx.x;
  const int m0 = blockIdx.y << 4, n0 = blockIdx.x << 6;
  const int ar = lane & 15, hi = lane >> 4, hsel = hi << 3;
  const f16* arow = A + (size_t)(m0 + ar) * DM;
  const f16* b0 = BT + (size_t)(n0 + ar) * DM;
  const f16* b1 = b0 + (size_t)16 * DM;
  const f16* b2 = b0 + (size_t)32 * DM;
  const f16* b3 = b0 + (size_t)48 * DM;
  v8f a0 = {}, a1 = {}, a2 = {}, a3 = {};
#pragma unroll
  for (int c = 0; c < DM; c += 32) {
    v16h av = ldp(arow, c + hsel);
    a0 = wmma_f16(av, ldp(b0, c + hsel), a0);
    a1 = wmma_f16(av, ldp(b1, c + hsel), a1);
    a2 = wmma_f16(av, ldp(b2, c + hsel), a2);
    a3 = wmma_f16(av, ldp(b3, c + hsel), a3);
  }
  F8 u[4];
  u[0].v = a0; u[1].v = a1; u[2].v = a2; u[3].v = a3;
#pragma unroll
  for (int r = 0; r < 8; ++r) {
    int mi = m0 + r + (hi << 3);
#pragma unroll
    for (int nn = 0; nn < 4; ++nn) {
      int col = n0 + (nn << 4) + ar;
      float val = u[nn].f[r];
      if (mode == 0) {
        ((f16*)out)[(size_t)mi * ldout + col] = (f16)val;
      } else if (mode == 1) {
        int b = mi >> 11, j = mi & (PSEQ - 1);
        int g = col >> 6, dd = col & 63;
        ((f16*)out)[(((size_t)b * NH + g) * DH + dd) * PSEQ + j] = (f16)val;
      } else {
        ((float*)out)[(size_t)mi * ldout + col] = val;
      }
    }
  }
}

// --------------------- fused talking-heads attention -----------------------
// Block: 256 threads = 8 waves; wave w handles mixed head w.
// Grid: B * (P/16); each block owns a 16-query tile.
__global__ __launch_bounds__(256) void k_attn(const f16* __restrict__ Qp,
                                              const f16* __restrict__ KH,
                                              const f16* __restrict__ VT,
                                              const float* __restrict__ W2,
                                              f16* __restrict__ Ob) {
  __shared__ f16 Kt[32 * DM];        // 32 keys x 512 features, shared by waves
  __shared__ f16 sb[NH][16 * 32];    // sigma exchange: [head][row][j]

  const int tid = threadIdx.x;
  const int w = tid >> 5, lane = tid & 31;
  const int b = blockIdx.x >> 7;
  const int ib = (blockIdx.x & 127) << 4;
  const int ar = lane & 15, hi = lane >> 4, hsel = hi << 3;

  // Q' A-tiles for this wave's head, resident in registers (16 K-chunks).
  V16 qa[16];
  {
    const f16* qrow = Qp + (size_t)(b * PSEQ + ib + ar) * 4096 + w * DM;
#pragma unroll
    for (int c = 0; c < 16; ++c) {
      qa[c].h2[0] = *(const v8h*)(qrow + c * 32 + hsel);
      qa[c].h2[1] = *(const v8h*)(qrow + c * 32 + 16 + hsel);
    }
  }
  float w2r[8];
#pragma unroll
  for (int h = 0; h < 8; ++h) w2r[h] = W2[h * 8 + w];

  float rm[8], ls[8];
#pragma unroll
  for (int r = 0; r < 8; ++r) { rm[r] = -1e30f; ls[r] = 0.0f; }

  const f16* kbase = KH + (size_t)b * PSEQ * DM;

  // ------------------ pass 1: per-row max & sum (online) ------------------
  for (int j0 = 0; j0 < PSEQ; j0 += 32) {
    __syncthreads();
    {
      const f16* src = kbase + (size_t)j0 * DM;
#pragma unroll
      for (int t = 0; t < 8; ++t)
        cp16_g2l(src + (size_t)(tid + t * 256) * 8, Kt + (size_t)(tid + t * 256) * 8);
    }
    if (j0 + 32 < PSEQ)
      __builtin_prefetch(kbase + (size_t)(j0 + 32) * DM + tid * 64, 0, 1);
    cp_wait();
    __syncthreads();

    v8f s0 = {}, s1 = {};
    const f16* kr0 = Kt + (size_t)ar * DM;
    const f16* kr1 = Kt + (size_t)(ar + 16) * DM;
#pragma unroll
    for (int c = 0; c < 16; ++c) {
      s0 = wmma_f16(qa[c].v, ldp(kr0, c * 32 + hsel), s0);
      s1 = wmma_f16(qa[c].v, ldp(kr1, c * 32 + hsel), s1);
    }
    F8 u0, u1; u0.v = s0; u1.v = s1;
#pragma unroll
    for (int r = 0; r < 8; ++r) {
      float nm = fmaxf(rm[r], fmaxf(u0.f[r], u1.f[r]));
      ls[r] = ls[r] * __expf(rm[r] - nm) + __expf(u0.f[r] - nm) + __expf(u1.f[r] - nm);
      rm[r] = nm;
    }
  }
  // merge per-lane (m,l) across the 16 lanes holding one row
#pragma unroll
  for (int r = 0; r < 8; ++r) {
#pragma unroll
    for (int off = 1; off < 16; off <<= 1) {
      float om = __shfl_xor(rm[r], off, 32);
      float ol = __shfl_xor(ls[r], off, 32);
      float nm = fmaxf(rm[r], om);
      ls[r] = ls[r] * __expf(rm[r] - nm) + ol * __expf(om - nm);
      rm[r] = nm;
    }
    ls[r] = 1.0f / ls[r];  // now exact 1/l
  }

  // ------------- pass 2: exact softmax, W2 mix, PV accumulate -------------
  v8f o0 = {}, o1 = {}, o2 = {}, o3 = {};
  const f16* vbase = VT + ((size_t)b * NH + w) * DH * PSEQ;

  for (int j0 = 0; j0 < PSEQ; j0 += 32) {
    __syncthreads();
    {
      const f16* src = kbase + (size_t)j0 * DM;
#pragma unroll
      for (int t = 0; t < 8; ++t)
        cp16_g2l(src + (size_t)(tid + t * 256) * 8, Kt + (size_t)(tid + t * 256) * 8);
    }
    cp_wait();
    __syncthreads();

    v8f s0 = {}, s1 = {};
    const f16* kr0 = Kt + (size_t)ar * DM;
    const f16* kr1 = Kt + (size_t)(ar + 16) * DM;
#pragma unroll
    for (int c = 0; c < 16; ++c) {
      s0 = wmma_f16(qa[c].v, ldp(kr0, c * 32 + hsel), s0);
      s1 = wmma_f16(qa[c].v, ldp(kr1, c * 32 + hsel), s1);
    }
    F8 u0, u1; u0.v = s0; u1.v = s1;
#pragma unroll
    for (int r = 0; r < 8; ++r) {  // C layout: row = r + 8*hi, col = lane&15
      int row = r + (hi << 3);
      sb[w][row * 32 + ar]      = (f16)(__expf(u0.f[r] - rm[r]) * ls[r]);
      sb[w][row * 32 + 16 + ar] = (f16)(__expf(u1.f[r] - rm[r]) * ls[r]);
    }
    __syncthreads();

    // A_mixed[i,j] = sum_h W2[h,w] * sigma_h[i,j], built directly in A layout
    V16 am;
#pragma unroll
    for (int part = 0; part < 2; ++part) {
      float accm[8] = {0, 0, 0, 0, 0, 0, 0, 0};
#pragma unroll
      for (int h = 0; h < 8; ++h) {
        v8h pv = *(const v8h*)(&sb[h][ar * 32 + part * 16 + hsel]);
#pragma unroll
        for (int e = 0; e < 8; ++e) accm[e] += w2r[h] * (float)pv[e];
      }
#pragma unroll
      for (int e = 0; e < 8; ++e) am.e[part * 8 + e] = (f16)accm[e];
    }

    // O += A_mixed[16x32] @ V[32x64]; VT rows are contiguous in j
    const f16* vr0 = vbase + (size_t)ar * PSEQ + j0;
    o0 = wmma_f16(am.v, ldp(vr0, hsel), o0);
    o1 = wmma_f16(am.v, ldp(vr0 + (size_t)16 * PSEQ, hsel), o1);
    o2 = wmma_f16(am.v, ldp(vr0 + (size_t)32 * PSEQ, hsel), o2);
    o3 = wmma_f16(am.v, ldp(vr0 + (size_t)48 * PSEQ, hsel), o3);
  }

  F8 q[4];
  q[0].v = o0; q[1].v = o1; q[2].v = o2; q[3].v = o3;
#pragma unroll
  for (int r = 0; r < 8; ++r) {
    size_t row = (size_t)(b * PSEQ + ib + r + (hi << 3)) * DM + w * DH + ar;
    Ob[row +  0] = (f16)q[0].f[r];
    Ob[row + 16] = (f16)q[1].f[r];
    Ob[row + 32] = (f16)q[2].f[r];
    Ob[row + 48] = (f16)q[3].f[r];
  }
}

// ------------------------------- launch ------------------------------------

extern "C" void kernel_launch(void* const* d_in, const int* in_sizes, int n_in,
                              void* d_out, int out_size, void* d_ws, size_t ws_size,
                              hipStream_t stream) {
  (void)in_sizes; (void)n_in; (void)out_size; (void)ws_size;
  const float* x  = (const float*)d_in[0];
  const float* Wq = (const float*)d_in[1];
  const float* Wk = (const float*)d_in[2];
  const float* Wv = (const float*)d_in[3];
  const float* W1 = (const float*)d_in[4];
  const float* W2 = (const float*)d_in[5];
  const float* Wp = (const float*)d_in[6];

  char* ws = (char*)d_ws;
  size_t off = 0;
  auto alloc = [&](size_t bytes) -> void* {
    void* p = ws + off;
    off += (bytes + 255) & ~(size_t)255;
    return p;
  };
  f16* XH   = (f16*)alloc((size_t)4096 * 512 * 2);   // x as f16
  f16* WqTb = (f16*)alloc((size_t)4096 * 512 * 2);   // pre-mixed Wq', [N,K]
  f16* WkT  = (f16*)alloc((size_t)512 * 512 * 2);
  f16* WvT  = (f16*)alloc((size_t)512 * 512 * 2);
  f16* WpT  = (f16*)alloc((size_t)512 * 512 * 2);
  f16* Qp   = (f16*)alloc((size_t)4096 * 4096 * 2);  // Q' [B*P, 8*512]
  f16* KH   = (f16*)alloc((size_t)4096 * 512 * 2);   // K  [B*P, 512]
  f16* VT   = (f16*)alloc((size_t)4096 * 512 * 2);   // V^T [b][g][64][P]
  f16* Ob   = (f16*)alloc((size_t)4096 * 512 * 2);   // attention out, f16

  k_cvt_f16<<<(4096 * 512 + 255) / 256, 256, 0, stream>>>(x, XH, 4096 * 512);
  k_prep_wq<<<(4096 * 512 + 255) / 256, 256, 0, stream>>>(Wq, W1, WqTb);
  k_transpose512<<<(512 * 512 + 255) / 256, 256, 0, stream>>>(Wk, WkT);
  k_transpose512<<<(512 * 512 + 255) / 256, 256, 0, stream>>>(Wv, WvT);
  k_transpose512<<<(512 * 512 + 255) / 256, 256, 0, stream>>>(Wp, WpT);

  k_gemm<<<dim3(64, 256), 32, 0, stream>>>(XH, WqTb, Qp, 4096, 0);  // Q'
  k_gemm<<<dim3(8, 256), 32, 0, stream>>>(XH, WkT, KH, 512, 0);     // K
  k_gemm<<<dim3(8, 256), 32, 0, stream>>>(XH, WvT, VT, 0, 1);       // V^T

  k_attn<<<256, 256, 0, stream>>>(Qp, KH, VT, W2, Ob);

  k_gemm<<<dim3(8, 256), 32, 0, stream>>>(Ob, WpT, d_out, 512, 2);  // proj
}
